// DiT_29394756174286
// MI455X (gfx1250) — compile-verified, matrix operand order checked by reading, not run on hardware
//
#include <hip/hip_runtime.h>
#include <math.h>

// ---------------- model constants ----------------
#define BB       8
#define LL       256
#define IN_CH    256
#define COND     2048
#define D_MODEL  768
#define DEPTH    12
#define D_STATE  16
#define D_CONV   4
#define D_INNER  1536
#define DT_RANK  48
#define N_HEADS  8
#define HEAD_DIM 96
#define L_TEXT   35
#define FREQ     256
#define NTOK     (BB * LL)                // 2048
#define XDBL_W   (DT_RANK + 2 * D_STATE)  // 80

typedef __attribute__((ext_vector_type(16))) _Float16 v16h;
typedef __attribute__((ext_vector_type(8)))  _Float16 v8h;
typedef __attribute__((ext_vector_type(8)))  float    v8f;
typedef __attribute__((ext_vector_type(4)))  float    v4f;

__device__ __forceinline__ float silu_f(float v) { return v / (1.f + __expf(-v)); }

__device__ __forceinline__ v4f vzero4() {
  v4f z; z[0] = 0.f; z[1] = 0.f; z[2] = 0.f; z[3] = 0.f; return z;
}

// =====================================================================
// WMMA GEMM:  C[M][N] = A[M][K](row-major, stride lda) * W[N][K]^T + bias
// 256 threads = 8 wave32; block tile 128x128; K-step 32.
// Wave grid 4(M) x 2(N); each wave computes a 32x64 patch:
// 2 A-frags x 4 B-frags -> 8 v_wmma per K-step.
// Software pipeline: tile k is cvt'd+stored to LDS while tile k+1's
// global loads are in flight (their wait lands next iteration), and a
// global_prefetch runs 2 tiles ahead. Rows are clamped (not guarded);
// K tail handled by a single uniform branch (only dt GEMM, K=48).
// =====================================================================
#define TBM 128
#define TBN 128
#define TBK 32
#define LDSK 40   // padded K stride (halves) to dodge bank conflicts

__global__ __launch_bounds__(256)
void wmma_gemm_k(const float* __restrict__ A, int lda,
                 const float* __restrict__ W,
                 const float* __restrict__ bias,
                 float* __restrict__ C,
                 int M, int N, int K)
{
  __shared__ _Float16 As[TBM][LDSK];
  __shared__ _Float16 Bs[TBN][LDSK];

  const int tid  = threadIdx.x;
  const int lane = tid & 31;
  const int wave = tid >> 5;
  const int wm   = wave & 3;     // 0..3  (M sub-tile, 32 rows each)
  const int wn   = wave >> 2;    // 0..1  (N sub-tile, 64 cols each)
  const int bm   = blockIdx.x * TBM;
  const int bn   = blockIdx.y * TBN;
  const int frow = lane & 15;    // fragment row (M or N within 16)
  const int grp  = lane >> 4;    // half-wave group for K interleave

  // staging: both tiles are 128 rows x 32 K; thread -> row tid>>1,
  // 16 consecutive K starting at (tid&1)*16
  const int srow = tid >> 1;
  const int scol = (tid & 1) * 16;
  int agr = bm + srow; if (agr > M - 1) agr = M - 1;
  int bgr = bn + srow; if (bgr > N - 1) bgr = N - 1;
  const float* aptr = A + (size_t)agr * lda + scol;
  const float* bptr = W + (size_t)bgr * K + scol;

  v4f aR[4], bR[4];   // register-staged tile (16 floats each of A,B)

  // ---- tile loader: uniform full/tail branch keeps hot path branch-free
  auto load_tile = [&](int k0) {
    if (k0 + TBK <= K) {
      const float* as = aptr + k0;
      const float* bs = bptr + k0;
#pragma unroll
      for (int c = 0; c < 4; ++c) {
        aR[c] = *(const v4f*)(as + c * 4);
        bR[c] = *(const v4f*)(bs + c * 4);
      }
    } else {            // K tail (K % 32 != 0, K % 8 == 0): zero-pad chunks
      const float* as = aptr + k0;
      const float* bs = bptr + k0;
      if (k0 + scol + 8 <= K) { aR[0] = *(const v4f*)as; aR[1] = *(const v4f*)(as + 4); }
      else                    { aR[0] = vzero4();        aR[1] = vzero4(); }
      if (k0 + scol + 16 <= K){ aR[2] = *(const v4f*)(as + 8); aR[3] = *(const v4f*)(as + 12); }
      else                    { aR[2] = vzero4();        aR[3] = vzero4(); }
      if (k0 + scol + 8 <= K) { bR[0] = *(const v4f*)bs; bR[1] = *(const v4f*)(bs + 4); }
      else                    { bR[0] = vzero4();        bR[1] = vzero4(); }
      if (k0 + scol + 16 <= K){ bR[2] = *(const v4f*)(bs + 8); bR[3] = *(const v4f*)(bs + 12); }
      else                    { bR[2] = vzero4();        bR[3] = vzero4(); }
    }
  };

  // ---- commit staged registers to LDS as f16 ----
  auto commit_tile = [&]() {
    v8h p, q;
#pragma unroll
    for (int i = 0; i < 4; ++i) { p[i] = (_Float16)aR[0][i]; p[i + 4] = (_Float16)aR[1][i]; }
#pragma unroll
    for (int i = 0; i < 4; ++i) { q[i] = (_Float16)aR[2][i]; q[i + 4] = (_Float16)aR[3][i]; }
    *(v8h*)&As[srow][scol]     = p;
    *(v8h*)&As[srow][scol + 8] = q;
#pragma unroll
    for (int i = 0; i < 4; ++i) { p[i] = (_Float16)bR[0][i]; p[i + 4] = (_Float16)bR[1][i]; }
#pragma unroll
    for (int i = 0; i < 4; ++i) { q[i] = (_Float16)bR[2][i]; q[i + 4] = (_Float16)bR[3][i]; }
    *(v8h*)&Bs[srow][scol]     = p;
    *(v8h*)&Bs[srow][scol + 8] = q;
  };

  v8f acc[2][4] = {};

  load_tile(0);
  for (int k0 = 0; k0 < K; k0 += TBK) {
    commit_tile();                              // cvt+store tile k (waits on its loads)
    if (k0 + TBK < K) load_tile(k0 + TBK);      // tile k+1 loads fly behind the WMMAs
    if (k0 + 2 * TBK < K) {                     // L2 prefetch 2 tiles ahead
      __builtin_prefetch(aptr + k0 + 2 * TBK, 0, 1);
      __builtin_prefetch(bptr + k0 + 2 * TBK, 0, 1);
    }
    __syncthreads();

    // fragments per ISA 16-bit layout:
    // grp0: K0..7 | K16..23 ; grp1: K8..15 | K24..31
    v16h af[2], bf[4];
#pragma unroll
    for (int i = 0; i < 2; ++i) {
      const int ar = wm * 32 + i * 16 + frow;
      const v8h alo = *(const v8h*)&As[ar][grp * 8];
      const v8h ahi = *(const v8h*)&As[ar][16 + grp * 8];
      af[i] = __builtin_shufflevector(alo, ahi,
          0,1,2,3,4,5,6,7,8,9,10,11,12,13,14,15);
    }
#pragma unroll
    for (int j = 0; j < 4; ++j) {
      const int br = wn * 64 + j * 16 + frow;
      const v8h blo = *(const v8h*)&Bs[br][grp * 8];
      const v8h bhi = *(const v8h*)&Bs[br][16 + grp * 8];
      bf[j] = __builtin_shufflevector(blo, bhi,
          0,1,2,3,4,5,6,7,8,9,10,11,12,13,14,15);
    }
#pragma unroll
    for (int i = 0; i < 2; ++i)
#pragma unroll
      for (int j = 0; j < 4; ++j)
        acc[i][j] = __builtin_amdgcn_wmma_f32_16x16x32_f16(
            false, af[i], false, bf[j], (short)0, acc[i][j], false, false);
    __syncthreads();
  }

  // ---- epilogue: D layout VGPR r -> M = r + grp*8, N = frow ----
  int   cj[4];
  float bj[4];
#pragma unroll
  for (int j = 0; j < 4; ++j) {
    cj[j] = bn + wn * 64 + j * 16 + frow;
    bj[j] = (bias && cj[j] < N) ? bias[cj[j]] : 0.f;
  }
#pragma unroll
  for (int i = 0; i < 2; ++i) {
    const int rbase = bm + wm * 32 + i * 16 + grp * 8;
#pragma unroll
    for (int r = 0; r < 8; ++r) {
      const int row = rbase + r;
      if (row < M) {
#pragma unroll
        for (int j = 0; j < 4; ++j)
          if (cj[j] < N) C[(size_t)row * N + cj[j]] = acc[i][j][r] + bj[j];
      }
    }
  }
}

// =====================================================================
// Naive GEMM for tiny M (conditioning path): C = act(A*W^T + bias)
// =====================================================================
__global__ void small_gemm_k(const float* __restrict__ A,
                             const float* __restrict__ W,
                             const float* __restrict__ bias,
                             float* __restrict__ C,
                             int M, int N, int K, int act)
{
  const int idx = blockIdx.x * blockDim.x + threadIdx.x;
  if (idx >= M * N) return;
  const int m = idx / N, n = idx % N;
  const float* a = A + (size_t)m * K;
  const float* w = W + (size_t)n * K;
  float s = bias ? bias[n] : 0.f;
  for (int k = 0; k < K; ++k) s += a[k] * w[k];
  if (act == 1) s = silu_f(s);
  C[idx] = s;
}

// =====================================================================
// LayerNorm + adaLN modulation: y = ln(x)*(1+scale[b]) + shift[b]
// =====================================================================
__global__ __launch_bounds__(256)
void ln_mod_k(const float* __restrict__ x,
              const float* __restrict__ shift,
              const float* __restrict__ scale, int cstride,
              float* __restrict__ y, int D, int l_per_b)
{
  __shared__ float rs[256];
  __shared__ float rq[256];
  const int row = blockIdx.x;
  const int b = row / l_per_b;
  const float* xr = x + (size_t)row * D;
  float s = 0.f, q = 0.f;
  for (int c = threadIdx.x; c < D; c += 256) {
    const float v = xr[c];
    s += v; q += v * v;
  }
  rs[threadIdx.x] = s; rq[threadIdx.x] = q;
  __syncthreads();
  for (int st = 128; st > 0; st >>= 1) {
    if (threadIdx.x < st) {
      rs[threadIdx.x] += rs[threadIdx.x + st];
      rq[threadIdx.x] += rq[threadIdx.x + st];
    }
    __syncthreads();
  }
  const float mean = rs[0] / (float)D;
  const float var  = rq[0] / (float)D - mean * mean;
  const float rinv = rsqrtf(var + 1e-6f);
  float* yr = y + (size_t)row * D;
  for (int c = threadIdx.x; c < D; c += 256) {
    float v = (xr[c] - mean) * rinv;
    if (scale)
      v = v * (1.f + scale[(size_t)b * cstride + c]) + shift[(size_t)b * cstride + c];
    yr[c] = v;
  }
}

// gated residual: x += gate[b]*y  (gate null -> plain add)
__global__ void resid_gate_k(float* __restrict__ x, const float* __restrict__ y,
                             const float* __restrict__ gate, int gstride,
                             int D, int l_per_b, int n)
{
  const int i = blockIdx.x * blockDim.x + threadIdx.x;
  if (i >= n) return;
  const int c = i % D;
  const int b = (i / D) / l_per_b;
  const float g = gate ? gate[(size_t)b * gstride + c] : 1.f;
  x[i] += g * y[i];
}

// causal depthwise conv (width 4) + bias + silu. input = u half of uz.
__global__ void conv_silu_k(const float* __restrict__ uz,
                            const float* __restrict__ cw,
                            const float* __restrict__ cb,
                            float* __restrict__ uc, int n)
{
  const int idx = blockIdx.x * blockDim.x + threadIdx.x;
  if (idx >= n) return;
  const int d = idx % D_INNER;
  const int l = (idx / D_INNER) % LL;
  const int b = idx / (D_INNER * LL);
  float s = cb[d];
#pragma unroll
  for (int j = 0; j < D_CONV; ++j) {
    const int ls = l - (D_CONV - 1) + j;
    if (ls >= 0)
      s += uz[((size_t)(b * LL + ls)) * (2 * D_INNER) + d] * cw[d * D_CONV + j];
  }
  uc[idx] = silu_f(s);
}

__global__ void softplus_k(float* __restrict__ v, int n)
{
  const int i = blockIdx.x * blockDim.x + threadIdx.x;
  if (i >= n) return;
  const float x = v[i];
  v[i] = (x > 20.f) ? x : log1pf(__expf(x));
}

__global__ void gelu_k(float* __restrict__ v, int n)
{
  const int i = blockIdx.x * blockDim.x + threadIdx.x;
  if (i >= n) return;
  const float x = v[i];
  v[i] = 0.5f * x * (1.f + erff(x * 0.70710678118f));
}

__global__ void silu_k(const float* __restrict__ a, float* __restrict__ o, int n)
{
  const int i = blockIdx.x * blockDim.x + threadIdx.x;
  if (i < n) o[i] = silu_f(a[i]);
}

__global__ void add_k(const float* __restrict__ a, const float* __restrict__ b,
                      float* __restrict__ o, int n)
{
  const int i = blockIdx.x * blockDim.x + threadIdx.x;
  if (i < n) o[i] = a[i] + b[i];
}

// fl_e += flpos (broadcast over batch)
__global__ void addpos_k(float* __restrict__ fle, const float* __restrict__ pos, int n)
{
  const int i = blockIdx.x * blockDim.x + threadIdx.x;
  if (i >= n) return;
  const int c = i % D_MODEL;
  const int lt = (i / D_MODEL) % L_TEXT;
  fle[i] += pos[(size_t)lt * D_MODEL + c];
}

// sinusoidal time embedding: [cos(t*f) | sin(t*f)], f_i = 10000^{-i/128}
__global__ void time_embed_k(const float* __restrict__ t, float* __restrict__ e)
{
  const int idx = blockIdx.x * blockDim.x + threadIdx.x;
  if (idx >= BB * FREQ) return;
  const int b = idx / FREQ, c = idx % FREQ;
  const float f = __expf(-logf(10000.f) * (float)(c & 127) / 128.f);
  const float ang = t[b] * f;
  e[idx] = (c < 128) ? __cosf(ang) : __sinf(ang);
}

// =====================================================================
// Selective scan: thread per (b,d); 16 states in registers, fused
// (y + u*Dp) * silu(z) epilogue.
// =====================================================================
__global__ void mamba_scan_k(const float* __restrict__ dt,
                             const float* __restrict__ uc,
                             const float* __restrict__ uz,    // z half
                             const float* __restrict__ xdbl,  // [tok][80]
                             const float* __restrict__ A_log, // [D_INNER][16]
                             const float* __restrict__ Dp,
                             float* __restrict__ y)
{
  const int idx = blockIdx.x * blockDim.x + threadIdx.x;
  if (idx >= BB * D_INNER) return;
  const int b = idx / D_INNER, d = idx % D_INNER;
  float A[D_STATE], h[D_STATE];
#pragma unroll
  for (int n = 0; n < D_STATE; ++n) {
    A[n] = -__expf(A_log[(size_t)d * D_STATE + n]);
    h[n] = 0.f;
  }
  const float Dd = Dp[d];
  for (int l = 0; l < LL; ++l) {
    const size_t r = (size_t)(b * LL + l);
    const float dtv = dt[r * D_INNER + d];
    const float uv  = uc[r * D_INNER + d];
    const float* xr = xdbl + r * XDBL_W;
    float acc = 0.f;
#pragma unroll
    for (int n = 0; n < D_STATE; ++n) {
      const float dA = __expf(dtv * A[n]);
      h[n] = dA * h[n] + dtv * xr[DT_RANK + n] * uv;
      acc += h[n] * xr[DT_RANK + D_STATE + n];
    }
    const float zv = uz[r * (2 * D_INNER) + D_INNER + d];
    y[r * D_INNER + d] = (acc + uv * Dd) * silu_f(zv);
  }
}

// =====================================================================
// Cross attention: block per (b,h); K/V (35x96) staged in LDS;
// thread per query token; scores fully register-resident.
// =====================================================================
__global__ __launch_bounds__(256)
void cross_attn_k(const float* __restrict__ qkv_x,   // [NTOK][2304], q cols 0..767
                  const float* __restrict__ qkv_fl,  // [B*35][2304], k@768, v@1536
                  float* __restrict__ out)           // [NTOK][768]
{
  __shared__ float Ks[L_TEXT][HEAD_DIM];
  __shared__ float Vs[L_TEXT][HEAD_DIM];
  const int b = blockIdx.x >> 3;
  const int h = blockIdx.x & 7;
  for (int i = threadIdx.x; i < L_TEXT * HEAD_DIM; i += 256) {
    const int kt = i / HEAD_DIM, dd = i % HEAD_DIM;
    const size_t row = (size_t)(b * L_TEXT + kt) * (3 * D_MODEL);
    Ks[kt][dd] = qkv_fl[row + D_MODEL + h * HEAD_DIM + dd];
    Vs[kt][dd] = qkv_fl[row + 2 * D_MODEL + h * HEAD_DIM + dd];
  }
  __syncthreads();

  const int l = threadIdx.x;                  // LL == 256 == blockDim
  const size_t qrow = (size_t)(b * LL + l) * (3 * D_MODEL) + h * HEAD_DIM;
  float s[L_TEXT];
#pragma unroll
  for (int kt = 0; kt < L_TEXT; ++kt) s[kt] = 0.f;
  for (int dd = 0; dd < HEAD_DIM; ++dd) {
    const float qv = qkv_x[qrow + dd];
#pragma unroll
    for (int kt = 0; kt < L_TEXT; ++kt) s[kt] += qv * Ks[kt][dd];
  }
  const float scale = 0.1020620726f;          // 1/sqrt(96)
  float mx = -1e30f;
#pragma unroll
  for (int kt = 0; kt < L_TEXT; ++kt) { s[kt] *= scale; mx = fmaxf(mx, s[kt]); }
  float sum = 0.f;
#pragma unroll
  for (int kt = 0; kt < L_TEXT; ++kt) { s[kt] = __expf(s[kt] - mx); sum += s[kt]; }
  const float inv = 1.f / sum;
  const size_t orow = (size_t)(b * LL + l) * D_MODEL + h * HEAD_DIM;
  for (int dd = 0; dd < HEAD_DIM; ++dd) {
    float o = 0.f;
#pragma unroll
    for (int kt = 0; kt < L_TEXT; ++kt) o += s[kt] * Vs[kt][dd];
    out[orow + dd] = o * inv;
  }
}

// =====================================================================
// Host orchestration
// =====================================================================
extern "C" void kernel_launch(void* const* d_in, const int* in_sizes, int n_in,
                              void* d_out, int out_size, void* d_ws, size_t ws_size,
                              hipStream_t stream)
{
  (void)in_sizes; (void)n_in; (void)out_size; (void)ws_size;
  const float* x_in   = (const float*)d_in[0];
  const float* t_in   = (const float*)d_in[1];
  const float* fc_in  = (const float*)d_in[2];
  const float* fl_in  = (const float*)d_in[3];
  const float* xw     = (const float*)d_in[4];
  const float* xb     = (const float*)d_in[5];
  const float* tw1    = (const float*)d_in[6];
  const float* tb1    = (const float*)d_in[7];
  const float* tw2    = (const float*)d_in[8];
  const float* tb2    = (const float*)d_in[9];
  const float* fcw1   = (const float*)d_in[10];
  const float* fcb1   = (const float*)d_in[11];
  const float* fcw2   = (const float*)d_in[12];
  const float* fcb2   = (const float*)d_in[13];
  const float* flw    = (const float*)d_in[14];
  const float* flb    = (const float*)d_in[15];
  const float* flpos  = (const float*)d_in[16];
  const float* ada_w  = (const float*)d_in[17];
  const float* ada_b  = (const float*)d_in[18];
  const float* in_w   = (const float*)d_in[19];
  const float* conv_w = (const float*)d_in[20];
  const float* conv_b = (const float*)d_in[21];
  const float* xprj_w = (const float*)d_in[22];
  const float* dt_w   = (const float*)d_in[23];
  const float* dt_b   = (const float*)d_in[24];
  const float* A_log  = (const float*)d_in[25];
  const float* Dp     = (const float*)d_in[26];
  const float* out_w  = (const float*)d_in[27];
  const float* qkv_w  = (const float*)d_in[28];
  const float* qkv_b  = (const float*)d_in[29];
  const float* ao_w   = (const float*)d_in[30];
  const float* ao_b   = (const float*)d_in[31];
  const float* fw1    = (const float*)d_in[32];
  const float* fb1    = (const float*)d_in[33];
  const float* fw2    = (const float*)d_in[34];
  const float* fb2    = (const float*)d_in[35];
  const float* fada_w = (const float*)d_in[36];
  const float* fada_b = (const float*)d_in[37];
  const float* fin_w  = (const float*)d_in[38];
  const float* fin_b  = (const float*)d_in[39];
  float* out = (float*)d_out;

  // workspace bump allocator (floats)
  float* ws = (float*)d_ws;
  size_t off = 0;
  auto alloc = [&](size_t n) { float* p = ws + off; off += n; return p; };
  float* xbuf  = alloc((size_t)NTOK * D_MODEL);
  float* xn    = alloc((size_t)NTOK * D_MODEL);
  float* tmb   = alloc((size_t)BB * FREQ);
  float* thid  = alloc((size_t)BB * D_MODEL);
  float* te    = alloc((size_t)BB * D_MODEL);
  float* fch   = alloc((size_t)BB * D_MODEL);
  float* fce   = alloc((size_t)BB * D_MODEL);
  float* cbuf  = alloc((size_t)BB * D_MODEL);
  float* scb   = alloc((size_t)BB * D_MODEL);
  float* modb  = alloc((size_t)BB * 6 * D_MODEL);
  float* fle   = alloc((size_t)BB * L_TEXT * D_MODEL);
  float* uz    = alloc((size_t)NTOK * 2 * D_INNER);
  float* ucv   = alloc((size_t)NTOK * D_INNER);
  float* xdb   = alloc((size_t)NTOK * XDBL_W);
  float* dtb   = alloc((size_t)NTOK * D_INNER);
  float* ybuf  = alloc((size_t)NTOK * D_INNER);
  float* tmp   = alloc((size_t)NTOK * D_MODEL);
  float* qkvx  = alloc((size_t)NTOK * 3 * D_MODEL);
  float* qkvf  = alloc((size_t)BB * L_TEXT * 3 * D_MODEL);
  float* abuf  = alloc((size_t)NTOK * D_MODEL);
  float* hbuf  = alloc((size_t)NTOK * 4 * D_MODEL);
  float* fmod  = alloc((size_t)BB * 2 * D_MODEL);

  auto gemm = [&](const float* A, int lda, const float* W, const float* bias,
                  float* C, int M, int N, int K) {
    dim3 g((M + TBM - 1) / TBM, (N + TBN - 1) / TBN);
    wmma_gemm_k<<<g, 256, 0, stream>>>(A, lda, W, bias, C, M, N, K);
  };
  auto sgemm = [&](const float* A, const float* W, const float* bias,
                   float* C, int M, int N, int K, int act) {
    small_gemm_k<<<((M * N) + 255) / 256, 256, 0, stream>>>(A, W, bias, C, M, N, K, act);
  };
  auto ew = [&](int n) { return dim3((n + 255) / 256); };

  // ---- input projection: x -> D_MODEL ----
  gemm(x_in, IN_CH, xw, xb, xbuf, NTOK, D_MODEL, IN_CH);

  // ---- conditioning path (tiny M) ----
  time_embed_k<<<ew(BB * FREQ), 256, 0, stream>>>(t_in, tmb);
  sgemm(tmb, tw1, tb1, thid, BB, D_MODEL, FREQ, 1);
  sgemm(thid, tw2, tb2, te, BB, D_MODEL, D_MODEL, 0);
  sgemm(fc_in, fcw1, fcb1, fch, BB, D_MODEL, COND, 1);
  sgemm(fch, fcw2, fcb2, fce, BB, D_MODEL, D_MODEL, 0);
  add_k<<<ew(BB * D_MODEL), 256, 0, stream>>>(te, fce, cbuf, BB * D_MODEL);
  silu_k<<<ew(BB * D_MODEL), 256, 0, stream>>>(cbuf, scb, BB * D_MODEL);

  // ---- text embedding ----
  sgemm(fl_in, flw, flb, fle, BB * L_TEXT, D_MODEL, COND, 0);
  addpos_k<<<ew(BB * L_TEXT * D_MODEL), 256, 0, stream>>>(fle, flpos, BB * L_TEXT * D_MODEL);

  for (int i = 0; i < DEPTH; ++i) {
    const float* inw_i  = in_w  + (size_t)i * 2 * D_INNER * D_MODEL;
    const float* cw_i   = conv_w + (size_t)i * D_INNER * D_CONV;
    const float* cb_i   = conv_b + (size_t)i * D_INNER;
    const float* xp_i   = xprj_w + (size_t)i * XDBL_W * D_INNER;
    const float* dtw_i  = dt_w  + (size_t)i * D_INNER * DT_RANK;
    const float* dtb_i  = dt_b  + (size_t)i * D_INNER;
    const float* al_i   = A_log + (size_t)i * D_INNER * D_STATE;
    const float* dp_i   = Dp    + (size_t)i * D_INNER;
    const float* ow_i   = out_w + (size_t)i * D_MODEL * D_INNER;
    const float* qw_i   = qkv_w + (size_t)i * 3 * D_MODEL * D_MODEL;
    const float* qb_i   = qkv_b + (size_t)i * 3 * D_MODEL;
    const float* aw_i   = ao_w  + (size_t)i * D_MODEL * D_MODEL;
    const float* ab_i   = ao_b  + (size_t)i * D_MODEL;
    const float* f1w_i  = fw1   + (size_t)i * 4 * D_MODEL * D_MODEL;
    const float* f1b_i  = fb1   + (size_t)i * 4 * D_MODEL;
    const float* f2w_i  = fw2   + (size_t)i * D_MODEL * 4 * D_MODEL;
    const float* f2b_i  = fb2   + (size_t)i * D_MODEL;

    // adaLN modulation: [B][6*768] = silu(c) @ ada_w[i]^T + ada_b[i]
    sgemm(scb, ada_w + (size_t)i * 6 * D_MODEL * D_MODEL,
          ada_b + (size_t)i * 6 * D_MODEL, modb, BB, 6 * D_MODEL, D_MODEL, 0);

    // ---- Mamba block ----
    ln_mod_k<<<NTOK, 256, 0, stream>>>(xbuf, modb + 0 * D_MODEL, modb + 1 * D_MODEL,
                                       6 * D_MODEL, xn, D_MODEL, LL);
    gemm(xn, D_MODEL, inw_i, nullptr, uz, NTOK, 2 * D_INNER, D_MODEL);
    conv_silu_k<<<ew(NTOK * D_INNER), 256, 0, stream>>>(uz, cw_i, cb_i, ucv, NTOK * D_INNER);
    gemm(ucv, D_INNER, xp_i, nullptr, xdb, NTOK, XDBL_W, D_INNER);
    gemm(xdb, XDBL_W, dtw_i, dtb_i, dtb, NTOK, D_INNER, DT_RANK);   // K=48 (tail-zeroed)
    softplus_k<<<ew(NTOK * D_INNER), 256, 0, stream>>>(dtb, NTOK * D_INNER);
    mamba_scan_k<<<ew(BB * D_INNER), 256, 0, stream>>>(dtb, ucv, uz, xdb, al_i, dp_i, ybuf);
    gemm(ybuf, D_INNER, ow_i, nullptr, tmp, NTOK, D_MODEL, D_INNER);
    resid_gate_k<<<ew(NTOK * D_MODEL), 256, 0, stream>>>(
        xbuf, tmp, modb + 2 * D_MODEL, 6 * D_MODEL, D_MODEL, LL, NTOK * D_MODEL);

    // ---- cross attention ----
    ln_mod_k<<<NTOK, 256, 0, stream>>>(xbuf, nullptr, nullptr, 0, xn, D_MODEL, LL);
    gemm(xn, D_MODEL, qw_i, qb_i, qkvx, NTOK, 3 * D_MODEL, D_MODEL);
    gemm(fle, D_MODEL, qw_i, qb_i, qkvf, BB * L_TEXT, 3 * D_MODEL, D_MODEL);
    cross_attn_k<<<BB * N_HEADS, 256, 0, stream>>>(qkvx, qkvf, abuf);
    gemm(abuf, D_MODEL, aw_i, ab_i, tmp, NTOK, D_MODEL, D_MODEL);
    resid_gate_k<<<ew(NTOK * D_MODEL), 256, 0, stream>>>(
        xbuf, tmp, nullptr, 0, D_MODEL, LL, NTOK * D_MODEL);

    // ---- MLP ----
    ln_mod_k<<<NTOK, 256, 0, stream>>>(xbuf, modb + 3 * D_MODEL, modb + 4 * D_MODEL,
                                       6 * D_MODEL, xn, D_MODEL, LL);
    gemm(xn, D_MODEL, f1w_i, f1b_i, hbuf, NTOK, 4 * D_MODEL, D_MODEL);
    gelu_k<<<ew(NTOK * 4 * D_MODEL), 256, 0, stream>>>(hbuf, NTOK * 4 * D_MODEL);
    gemm(hbuf, 4 * D_MODEL, f2w_i, f2b_i, tmp, NTOK, D_MODEL, 4 * D_MODEL);
    resid_gate_k<<<ew(NTOK * D_MODEL), 256, 0, stream>>>(
        xbuf, tmp, modb + 5 * D_MODEL, 6 * D_MODEL, D_MODEL, LL, NTOK * D_MODEL);
  }

  // ---- final adaLN + projection to IN_CH ----
  sgemm(scb, fada_w, fada_b, fmod, BB, 2 * D_MODEL, D_MODEL, 0);
  ln_mod_k<<<NTOK, 256, 0, stream>>>(xbuf, fmod + 0 * D_MODEL, fmod + 1 * D_MODEL,
                                     2 * D_MODEL, xn, D_MODEL, LL);
  gemm(xn, D_MODEL, fin_w, fin_b, out, NTOK, IN_CH, D_MODEL);
}
